// RGBuvHistBlock_38895223832965
// MI455X (gfx1250) — matrix operand order repaired and checked by reading, not compiled, not run on hardware
//
#include <hip/hip_runtime.h>
#include <hip/hip_bf16.h>

typedef __attribute__((ext_vector_type(16))) _Float16 v16h;
typedef __attribute__((ext_vector_type(8)))  float    v8f;

#define H_BINS 64
#define NPIX   160000          // 400*400
#define NBATCH 8
#define CHUNK  2048            // pixels per block
#define SUB    256             // pixels staged in LDS per pass

// bins*inv_sigma: bin_j*50 = -150 + j*(300/63)
__device__ __forceinline__ float bin50(int j) {
    return -150.0f + (float)j * (300.0f / 63.0f);
}

__device__ __forceinline__ float iq_kern(float d50, float b50) {
    float t = d50 - b50;
    return __builtin_amdgcn_rcpf(__builtin_fmaf(t, t, 1.0f));
}

__global__ void zero_ws_kernel(float* __restrict__ p, int n) {
    int i = blockIdx.x * blockDim.x + threadIdx.x;
    if (i < n) p[i] = 0.0f;
}

__global__ __launch_bounds__(256) void
hist_wmma_kernel(const float* __restrict__ x, const float* __restrict__ m,
                 float* __restrict__ acc)
{
    const int chunk = blockIdx.x;
    const int pair  = blockIdx.y;   // 0..2
    const int b     = blockIdx.z;   // 0..7
    const int tid   = threadIdx.x;
    const int lane  = tid & 31;
    const int wave  = tid >> 5;     // 8 waves
    const bool hi   = lane >= 16;

    // LDS staging: (du*50, wgt) pairs and dv*50, 16B-aligned for b128 loads
    __shared__ __align__(16) float2 s_uw[SUB];
    __shared__ __align__(16) float  s_dv[SUB];

    const float* xb = x + (size_t)b * 3 * NPIX;
    const float* mb = m + (size_t)b * 3 * NPIX;

    // wave -> output tiles: A-tile row tu, two B-tile cols tvbase, tvbase+1
    const int tu     = wave & 3;
    const int tvbase = (wave >> 2) * 2;

    const int mrow  = lane & 15;
    const float binA  = bin50(tu * 16 + mrow);
    const float binB0 = bin50(tvbase * 16 + mrow);
    const float binB1 = bin50((tvbase + 1) * 16 + mrow);

    v8f acc0 = {};
    v8f acc1 = {};

    const int base0 = chunk * CHUNK;
    for (int sc = 0; sc < CHUNK; sc += SUB) {
        // ---- stage SUB pixels into LDS ----
        int p = base0 + sc + tid;
        float du = 0.0f, dv = 0.0f, wg = 0.0f;
        if (p < NPIX) {
            float r  = __saturatef(xb[0 * NPIX + p]);
            float g  = __saturatef(xb[1 * NPIX + p]);
            float bl = __saturatef(xb[2 * NPIX + p]);
            float w  = (mb[p] > 0.5f) ? 1.0f : 0.0f;
            float iy = __fsqrt_rn(__builtin_fmaf(r, r,
                          __builtin_fmaf(g, g, __builtin_fmaf(bl, bl, 1e-6f))));
            float l0 = __logf(r  + 1e-6f);
            float l1 = __logf(g  + 1e-6f);
            float l2 = __logf(bl + 1e-6f);
            float la, lb, lc;
            if (pair == 0)      { la = l0; lb = l1; lc = l2; }
            else if (pair == 1) { la = l1; lb = l0; lc = l2; }
            else                { la = l2; lb = l0; lc = l1; }
            du = (la - lb) * 50.0f;
            dv = (la - lc) * 50.0f;
            wg = iy * w;
        }
        __syncthreads();                      // protect LDS from prior K-loop reads
        s_uw[tid] = make_float2(du, wg);
        s_dv[tid] = dv;
        __syncthreads();

        const float4* s_uw4 = reinterpret_cast<const float4*>(s_uw);
        const float4* s_dv4 = reinterpret_cast<const float4*>(s_dv);

        // ---- K-steps of 32 pixels ----
        for (int k = 0; k < SUB; k += 32) {
            // A fragment: 16-bit A 16x32 layout.
            // lane<16: K = {0..7, 16..23}; lane>=16: K = {8..15, 24..31}
            v16h a;
#pragma unroll
            for (int g = 0; g < 2; ++g) {
                int pstart = k + (hi ? 8 : 0) + g * 16;   // 8 consecutive pixels
#pragma unroll
                for (int q = 0; q < 4; ++q) {
                    float4 t4 = s_uw4[(pstart >> 1) + q]; // 2 pixels: (du,wg,du,wg)
                    a[g * 8 + 2 * q]     = (_Float16)(t4.y * iq_kern(t4.x, binA));
                    a[g * 8 + 2 * q + 1] = (_Float16)(t4.w * iq_kern(t4.z, binA));
                }
            }
            // B fragments: 32x16, lanes 0-15 hold K={0..7,16..23}, lanes 16-31 K={8..15,24..31}
            v16h b0, b1;
            int kb = k + (hi ? 8 : 0);
#pragma unroll
            for (int q = 0; q < 4; ++q) {
                float4 d4 = s_dv4[(kb >> 2) + (q & 1) + (q >> 1) * 4];         // 4 consecutive pixels
#pragma unroll
                for (int j = 0; j < 4; ++j) {
                    float d = (j == 0) ? d4.x : (j == 1) ? d4.y : (j == 2) ? d4.z : d4.w;
                    b0[4 * q + j] = (_Float16)iq_kern(d, binB0);
                    b1[4 * q + j] = (_Float16)iq_kern(d, binB1);
                }
            }
            acc0 = __builtin_amdgcn_wmma_f32_16x16x32_f16(
                       false, a, false, b0, (short)0, acc0, false, false);
            acc1 = __builtin_amdgcn_wmma_f32_16x16x32_f16(
                       false, a, false, b1, (short)0, acc1, false, false);
        }
    }

    // ---- merge partial 64x64 tile into global accumulator ----
    // D layout: VGPR i -> row i (lanes 0-15) / row 8+i (lanes 16-31); col = lane&15
    float* accp = acc + (size_t)(b * 3 + pair) * H_BINS * H_BINS;
    const int ncol = lane & 15;
#pragma unroll
    for (int i = 0; i < 8; ++i) {
        int mr = tu * 16 + i + (hi ? 8 : 0);
        unsafeAtomicAdd(&accp[mr * H_BINS + tvbase * 16 + ncol],       acc0[i]);
        unsafeAtomicAdd(&accp[mr * H_BINS + (tvbase + 1) * 16 + ncol], acc1[i]);
    }
}

__global__ __launch_bounds__(256) void
normalize_kernel(const float* __restrict__ acc, float* __restrict__ out)
{
    const int b = blockIdx.x;
    const int n = 3 * H_BINS * H_BINS;           // 12288 per batch
    const float* ab = acc + (size_t)b * n;
    float*       ob = out + (size_t)b * n;

    __shared__ float red[256];
    float s = 0.0f;
    for (int i = threadIdx.x; i < n; i += 256) s += ab[i];
    red[threadIdx.x] = s;
    __syncthreads();
    for (int st = 128; st > 0; st >>= 1) {
        if (threadIdx.x < st) red[threadIdx.x] += red[threadIdx.x + st];
        __syncthreads();
    }
    float inv = 1.0f / (red[0] + 1e-6f);
    for (int i = threadIdx.x; i < n; i += 256) ob[i] = ab[i] * inv;
}

extern "C" void kernel_launch(void* const* d_in, const int* in_sizes, int n_in,
                              void* d_out, int out_size, void* d_ws, size_t ws_size,
                              hipStream_t stream)
{
    const float* x = (const float*)d_in[0];   // (8,3,400,400) f32
    const float* m = (const float*)d_in[1];   // (8,3,400,400) f32
    float* out = (float*)d_out;               // (8,3,64,64)  f32
    float* acc = (float*)d_ws;                // 8*3*64*64 f32 = 384 KB scratch

    const int acc_n = NBATCH * 3 * H_BINS * H_BINS;   // 98304
    zero_ws_kernel<<<(acc_n + 255) / 256, 256, 0, stream>>>(acc, acc_n);

    const int chunks = (NPIX + CHUNK - 1) / CHUNK;    // 79
    dim3 grid(chunks, 3, NBATCH);                     // 1896 blocks
    hist_wmma_kernel<<<grid, 256, 0, stream>>>(x, m, acc);

    normalize_kernel<<<NBATCH, 256, 0, stream>>>(acc, out);
}
